// NeuronToSpatialGrid_19490561589315
// MI455X (gfx1250) — matrix operand order repaired and checked by reading, not compiled, not run on hardware
//
#include <hip/hip_runtime.h>

typedef __attribute__((ext_vector_type(16))) _Float16 v16h;
typedef __attribute__((ext_vector_type(8)))  _Float16 v8h;
typedef __attribute__((ext_vector_type(2)))  _Float16 v2h;
typedef __attribute__((ext_vector_type(8)))  float    v8f;

#define NB 4096        // neurons per batch
#define EB 256         // embedding dim
#define PB 4096        // grid points (64*64)
#define KT 32          // K tile (WMMA f16 K)
#define MB 64          // grid rows per block
#define NCHUNK (NB / KT)
#define SIGMA2 0.02f
#define REPS 1e-8f
#define WROW 40        // half stride of sW rows  (80 B, 16B aligned, conflict-padded)
#define FROW 40        // half stride of sF cols

__global__ __launch_bounds__(256)
void n2g_kernel(const float* __restrict__ feat,
                const float* __restrict__ pos,
                float* __restrict__ out)
{
    __shared__ __align__(16) _Float16 sW[MB * WROW];   // W tile, row-major [64][40]
    __shared__ __align__(16) _Float16 sF[EB * FROW];   // F chunk transposed [e][k]
    __shared__ float sRow[MB];

    const int tid  = threadIdx.x;
    const int lane = tid & 31;
    const int wave = tid >> 5;
    const int b    = blockIdx.y;
    const int rowbase = blockIdx.x * MB;

    // --- W-generation assignment: thread -> (row, 8-col strip) ---
    const int wrow = tid >> 2;            // 0..63
    const int wcol = (tid & 3) * 8;       // 0,8,16,24
    const int p    = rowbase + wrow;
    const float gx = (float)(p >> 6) * (1.0f / 63.0f);   // linspace(0,1,64), 'ij'
    const float gy = (float)(p & 63) * (1.0f / 63.0f);

    // --- F-load assignment: thread -> (k-pair, 16-wide e-strip) ---
    const int fk = (tid >> 4) * 2;        // k = fk, fk+1   (0,2,..,30)
    const int fe = (tid & 15) * 16;       // e strip 0..240

    // --- WMMA tile assignment: 8 waves cover 4 row-tiles x 16 col-tiles ---
    const int rt  = wave & 3;             // row tile (16 rows)
    const int cg  = wave >> 2;            // col half (128 cols)
    const int l16 = lane & 15;
    const int hi  = lane >> 4;            // 0/1: lane half

    v8f acc[8];
#pragma unroll
    for (int j = 0; j < 8; ++j) acc[j] = (v8f)0.0f;

    float psum = 0.0f;                    // f32 partial row-sum (this thread's strip)

    const float2* __restrict__ posB  = (const float2*)pos + (size_t)b * NB;
    const float*  __restrict__ featB = feat + (size_t)b * NB * EB;

    for (int kc = 0; kc < NCHUNK; ++kc) {
        const int n0 = kc * KT;

        // (1) compute 64x32 RBF weight tile -> f16 LDS, f32 row-sum partials
        {
            const float2* q = posB + n0 + wcol;
            _Float16 wh[8];
#pragma unroll
            for (int j = 0; j < 8; ++j) {
                float2 pq = q[j];
                float dx = gx - pq.x;
                float dy = gy - pq.y;
                float d2 = dx * dx + dy * dy;
                float w  = __expf(-d2 * (1.0f / SIGMA2));
                psum += w;
                wh[j] = (_Float16)w;
            }
            *(v8h*)(&sW[wrow * WROW + wcol]) = *(v8h*)wh;  // one ds_store_b128
        }

        // (2) load 2x16 f32 feature strip per thread, pack half2 (k,k+1) -> LDS
        {
            const float4* f0 = (const float4*)(featB + (size_t)(n0 + fk    ) * EB + fe);
            const float4* f1 = (const float4*)(featB + (size_t)(n0 + fk + 1) * EB + fe);
            float4 r0[4], r1[4];
#pragma unroll
            for (int v = 0; v < 4; ++v) { r0[v] = f0[v]; r1[v] = f1[v]; }
#pragma unroll
            for (int v = 0; v < 4; ++v) {
                const float a0[4] = { r0[v].x, r0[v].y, r0[v].z, r0[v].w };
                const float a1[4] = { r1[v].x, r1[v].y, r1[v].z, r1[v].w };
#pragma unroll
                for (int u = 0; u < 4; ++u) {
                    const int e = fe + v * 4 + u;
                    v2h pk = { (_Float16)a0[u], (_Float16)a1[u] };
                    *(v2h*)(&sF[e * FROW + fk]) = pk;      // one ds_store_b32
                }
            }
        }

        __syncthreads();

        // (3) WMMA: A = W rows (16x32), B = F (32x16) per col tile
        {
            union { v16h v; v8h h[2]; } A;
            const int mrow = rt * 16 + l16;
            const int ka   = hi * 8;                 // lane<16: K0..7/K16..23; else K8..15/K24..31
            A.h[0] = *(const v8h*)(&sW[mrow * WROW + ka]);
            A.h[1] = *(const v8h*)(&sW[mrow * WROW + 16 + ka]);

#pragma unroll
            for (int j = 0; j < 8; ++j) {
                const int e = cg * 128 + j * 16 + l16;   // B column per lane
                union { v16h v; v8h h[2]; } Bf;
                const _Float16* src = &sF[e * FROW + hi * 8]; // lanes 0-15: K0..7/K16..23; 16-31: K8..15/K24..31
                Bf.h[0] = *(const v8h*)(src);
                Bf.h[1] = *(const v8h*)(src + 16);
                acc[j] = __builtin_amdgcn_wmma_f32_16x16x32_f16(
                    false, A.v, false, Bf.v, (short)0, acc[j], false, false);
            }
        }

        __syncthreads();   // protect LDS before next chunk overwrites
    }

    // ---- reduce row sums (4 partials per row) ----
    if (tid < MB) sRow[tid] = REPS;
    __syncthreads();
    atomicAdd(&sRow[wrow], psum);          // ds_add_f32
    __syncthreads();

    // ---- deferred normalization + store (out[b][e][p], p fastest) ----
    const int mbase = rt * 16 + hi * 8;    // this lane's 8 consecutive M (= p) values
    float rinv[8];
#pragma unroll
    for (int r = 0; r < 8; ++r) rinv[r] = 1.0f / sRow[mbase + r];

#pragma unroll
    for (int j = 0; j < 8; ++j) {
        const int e = cg * 128 + j * 16 + l16;
        float* op = out + ((size_t)b * EB + e) * PB + rowbase + mbase;
        float4 o0 = { acc[j][0] * rinv[0], acc[j][1] * rinv[1],
                      acc[j][2] * rinv[2], acc[j][3] * rinv[3] };
        float4 o1 = { acc[j][4] * rinv[4], acc[j][5] * rinv[5],
                      acc[j][6] * rinv[6], acc[j][7] * rinv[7] };
        *(float4*)(op + 0) = o0;           // global_store_b128
        *(float4*)(op + 4) = o1;
    }
}

extern "C" void kernel_launch(void* const* d_in, const int* in_sizes, int n_in,
                              void* d_out, int out_size, void* d_ws, size_t ws_size,
                              hipStream_t stream) {
    const float* feat = (const float*)d_in[0];   // [4, 4096, 256] f32
    const float* pos  = (const float*)d_in[1];   // [4, 4096, 2]  f32
    float* out = (float*)d_out;                  // [4, 256, 64, 64] f32

    dim3 grid(PB / MB, 4);   // 64 row-blocks x 4 batches
    n2g_kernel<<<grid, 256, 0, stream>>>(feat, pos, out);
}
